// Scaled_Dot_Product_Attention_34308198760554
// MI455X (gfx1250) — matrix operand order repaired and checked by reading, not compiled, not run on hardware
//
#include <hip/hip_runtime.h>

typedef __attribute__((ext_vector_type(16))) _Float16 v16h;
typedef __attribute__((ext_vector_type(8)))  _Float16 v8h;
typedef __attribute__((ext_vector_type(4)))  _Float16 v4h;
typedef __attribute__((ext_vector_type(8)))  float    v8f;
typedef __attribute__((ext_vector_type(4)))  float    v4f;
typedef __attribute__((ext_vector_type(4)))  unsigned int v4u;
typedef __attribute__((ext_vector_type(8)))  int      v8i;
typedef __attribute__((ext_vector_type(4)))  int      v4i;

#define SLEN   2048
#define DHEAD  64
#define QTILE  64
#define KTILE  64
#define NWAVES 4
#define NBH    64            // B*H

#define SCALE_LOG2E 0.1803368801111204f   // (1/sqrt(64)) * log2(e)
#define NEGBIG      -1.0e30f

#define KH_BYTES ((size_t)NBH * SLEN * DHEAD * 2)   // f16 K copy
#define WS_NEEDED (2 * KH_BYTES)                    // + f16 V^T copy

// ---- 16-lane (half-wave) butterfly reductions via DPP (VALU pipe, no LDS) ----
template <int CTRL>
__device__ __forceinline__ float dpp_mov(float x) {
  return __builtin_bit_cast(float,
      __builtin_amdgcn_update_dpp(0, __builtin_bit_cast(int, x),
                                  CTRL, 0xf, 0xf, true));
}
__device__ __forceinline__ float row16_max(float x) {
  x = fmaxf(x, dpp_mov<0xB1>(x));    // quad_perm [1,0,3,2] : xor 1
  x = fmaxf(x, dpp_mov<0x4E>(x));    // quad_perm [2,3,0,1] : xor 2
  x = fmaxf(x, dpp_mov<0x141>(x));   // row_half_mirror
  x = fmaxf(x, dpp_mov<0x140>(x));   // row_mirror
  return x;
}
__device__ __forceinline__ float row16_sum(float x) {
  x += dpp_mov<0xB1>(x);
  x += dpp_mov<0x4E>(x);
  x += dpp_mov<0x141>(x);
  x += dpp_mov<0x140>(x);
  return x;
}

// Online-softmax row update (log2 domain; MASKED is compile-time).
#define SOFTMAX_ROWS(MASKED)                                                  \
  _Pragma("unroll")                                                           \
  for (int r = 0; r < 8; ++r) {                                               \
    float s0 = sc[0][r], s1 = sc[1][r], s2 = sc[2][r], s3 = sc[3][r];         \
    if (MASKED) {                                                             \
      const int qr = qrow0 + r + 8 * laneHi;                                  \
      if (kbase + laneLo > qr)      s0 = NEGBIG;                              \
      if (kbase + 16 + laneLo > qr) s1 = NEGBIG;                              \
      if (kbase + 32 + laneLo > qr) s2 = NEGBIG;                              \
      if (kbase + 48 + laneLo > qr) s3 = NEGBIG;                              \
    }                                                                         \
    const float tmax = row16_max(fmaxf(fmaxf(s0, s1), fmaxf(s2, s3)));        \
    const float mnew = fmaxf(mrow[r], tmax);                                  \
    const float corr = __builtin_amdgcn_exp2f(mrow[r] - mnew);                \
    const float p0 = __builtin_amdgcn_exp2f(s0 - mnew);                       \
    const float p1 = __builtin_amdgcn_exp2f(s1 - mnew);                       \
    const float p2 = __builtin_amdgcn_exp2f(s2 - mnew);                       \
    const float p3 = __builtin_amdgcn_exp2f(s3 - mnew);                       \
    const float rsum = row16_sum((p0 + p1) + (p2 + p3));                      \
    lrow[r] = lrow[r] * corr + rsum;                                          \
    mrow[r] = mnew;                                                           \
    Oacc[0][r] *= corr; Oacc[1][r] *= corr;                                   \
    Oacc[2][r] *= corr; Oacc[3][r] *= corr;                                   \
    v4h pk;                                                                   \
    pk[0] = (_Float16)p0; pk[1] = (_Float16)p1;                               \
    pk[2] = (_Float16)p2; pk[3] = (_Float16)p3;                               \
    *(v4h*)&sPw[(r + 8 * laneHi) * KTILE + 4 * laneLo] = pk;                  \
  }

// ---- TDM descriptor build + issue (ISA 8.3/8.4; data_size=3 -> 8B units) ----
__device__ __forceinline__ void tdm_load_2d(const void* lds_dst, const void* gsrc,
                                            unsigned tile_d0_u, unsigned tile_d1,
                                            unsigned tensor_d0_u, unsigned tensor_d1,
                                            unsigned stride0_u) {
  const unsigned long long ga = (unsigned long long)(size_t)gsrc;
  const unsigned lds_addr = (unsigned)(size_t)lds_dst;  // low 32b of generic = LDS offset
  v4u g0;
  g0[0] = 1u;                                            // count=1, user mode
  g0[1] = lds_addr;
  g0[2] = (unsigned)(ga & 0xFFFFFFFFu);
  g0[3] = (unsigned)((ga >> 32) & 0x01FFFFFFu) | (2u << 30);  // type=2 ("image")
  v8i g1;
  g1[0] = (int)(3u << 16);                               // wg_mask=0, data_size=8B
  g1[1] = (int)((tensor_d0_u & 0xFFFFu) << 16);          // tensor_dim0[15:0]
  g1[2] = (int)(((tensor_d1 & 0xFFFFu) << 16) | (tensor_d0_u >> 16));
  g1[3] = (int)(((tile_d0_u & 0xFFFFu) << 16) | (tensor_d1 >> 16));
  g1[4] = (int)(tile_d1 & 0xFFFFu);                      // tile_dim1 (tile_dim2=0)
  g1[5] = (int)stride0_u;                                // tensor_dim0_stride[31:0]
  g1[6] = 0;                                             // stride0[47:32], stride1[15:0]
  g1[7] = 0;                                             // stride1[47:16]
  v4i z = {0, 0, 0, 0};
#if defined(__clang_major__) && __clang_major__ >= 23
  v8i z8 = {0, 0, 0, 0, 0, 0, 0, 0};
  __builtin_amdgcn_tensor_load_to_lds(g0, g1, z, z, z8, 0);
#else
  __builtin_amdgcn_tensor_load_to_lds(g0, g1, z, z, 0);
#endif
}

// =====================  pre-pass: K->f16, V->f16 transposed+pi  =============
__global__ __launch_bounds__(128)
void prep_kv(const float* __restrict__ Vg, const float* __restrict__ Kg,
             _Float16* __restrict__ Kh, _Float16* __restrict__ Vt) {
  __shared__ __align__(32) _Float16 sT[DHEAD * KTILE];   // [d][pi(key)]
  const int tid  = threadIdx.x;
  const int srow = tid >> 1;                 // 0..63
  const int scol = (tid & 1) * 32;           // 0 or 32
  const int vcol = 4 * (srow & 15) + (srow >> 4);        // pi(srow)
  const int s0   = blockIdx.x * KTILE;
  const int bh   = blockIdx.y;

  const float* kp = Kg + ((size_t)bh * SLEN + s0 + srow) * DHEAD + scol;
  const float* vp = Vg + ((size_t)bh * SLEN + s0 + srow) * DHEAD + scol;
  _Float16*    ko = Kh + ((size_t)bh * SLEN + s0 + srow) * DHEAD + scol;

  #pragma unroll
  for (int c = 0; c < 4; ++c) {              // K: convert, same layout
    v4f x0 = *(const v4f*)(kp + 8 * c);
    v4f x1 = *(const v4f*)(kp + 8 * c + 4);
    v8h h;
    #pragma unroll
    for (int i = 0; i < 4; ++i) { h[i] = (_Float16)x0[i]; h[4 + i] = (_Float16)x1[i]; }
    *(v8h*)(ko + 8 * c) = h;
  }
  #pragma unroll
  for (int c = 0; c < 8; ++c) {              // V: transpose (+pi) via LDS
    v4f x = *(const v4f*)(vp + 4 * c);
    #pragma unroll
    for (int i = 0; i < 4; ++i)
      sT[(scol + 4 * c + i) * KTILE + vcol] = (_Float16)x[i];
  }
  __syncthreads();
  // dump: row d = srow, key-half = scol, coalesced b128 stores
  _Float16* vo = Vt + ((size_t)bh * DHEAD + srow) * SLEN + s0 + scol;
  #pragma unroll
  for (int c = 0; c < 4; ++c)
    *(v8h*)(vo + 8 * c) = *(const v8h*)&sT[srow * KTILE + scol + 8 * c];
}

// =====================  main: TDM-fed flash attention  ======================
__global__ __launch_bounds__(128)
void fa_fwd_tdm(const float* __restrict__ Qg, const _Float16* __restrict__ Kh,
                const _Float16* __restrict__ Vt, float* __restrict__ Og) {
  __shared__ __align__(32) _Float16 sK[2][KTILE * DHEAD];    // [key][d]
  __shared__ __align__(32) _Float16 sV[2][DHEAD * KTILE];    // [d][pi(key)]
  __shared__ __align__(32) _Float16 sP[NWAVES][16 * KTILE];  // [row][pi(key)]

  const int tid    = threadIdx.x;
  const int wave   = tid >> 5;
  const int lane   = tid & 31;
  const int laneLo = lane & 15;
  const int laneHi = lane >> 4;

  const int q0 = blockIdx.x * QTILE;
  const int bh = blockIdx.y;
  const float*    Qb  = Qg + (size_t)bh * SLEN * DHEAD;
  float*          Ob  = Og + (size_t)bh * SLEN * DHEAD;
  const _Float16* Kbh = Kh + (size_t)bh * SLEN * DHEAD;
  const _Float16* Vbh = Vt + (size_t)bh * DHEAD * SLEN;

  const int qrow0 = q0 + wave * 16;
  _Float16* sPw = &sP[wave][0];

  // ---- Q A-fragments with scale*log2e folded ----
  v16h Qa[2];
  {
    const float* qrow = Qb + (size_t)(qrow0 + laneLo) * DHEAD;
    #pragma unroll
    for (int f = 0; f < 2; ++f) {
      const int dbase = 32 * f + laneHi * 16;
      v4f a0 = *(const v4f*)(qrow + dbase);
      v4f a1 = *(const v4f*)(qrow + dbase + 4);
      v4f b0 = *(const v4f*)(qrow + dbase + 8);
      v4f b1 = *(const v4f*)(qrow + dbase + 12);
      #pragma unroll
      for (int i = 0; i < 4; ++i) {
        Qa[f][i]      = (_Float16)(a0[i] * SCALE_LOG2E);
        Qa[f][4 + i]  = (_Float16)(a1[i] * SCALE_LOG2E);
        Qa[f][8 + i]  = (_Float16)(b0[i] * SCALE_LOG2E);
        Qa[f][12 + i] = (_Float16)(b1[i] * SCALE_LOG2E);
      }
    }
  }

  v8f Oacc[4];
  #pragma unroll
  for (int j = 0; j < 4; ++j)
    #pragma unroll
    for (int i = 0; i < 8; ++i) Oacc[j][i] = 0.0f;

  float mrow[8], lrow[8];
  #pragma unroll
  for (int r = 0; r < 8; ++r) { mrow[r] = NEGBIG; lrow[r] = 0.0f; }

  const int nTiles = blockIdx.x + 1;   // causal: keys 0 .. q0+63

  // ---- prologue: async-stage tile 0 into buffer 0 (wave 0 issues) ----
  if (wave == 0) {
    // K tile: 64 contiguous rows of 128B = one 8KB line (1 x 1024 units)
    tdm_load_2d(&sK[0][0], Kbh, 1024, 1, 32768, 1, 1024);
    // V^T tile: 64 rows x 128B, row stride = SLEN*2 bytes = 512 units
    tdm_load_2d(&sV[0][0], Vbh, 16, 64, 512, 64, 512);
  }

  for (int kt = 0; kt < nTiles; ++kt) {
    const int cur   = kt & 1;
    const int kbase = kt * KTILE;

    if (wave == 0) {
      if (kt + 1 < nTiles) {           // pipeline next tile into other buffer
        const int nb = cur ^ 1;
        const size_t koff = (size_t)(kbase + KTILE) * DHEAD;
        tdm_load_2d(&sK[nb][0], Kbh + koff, 1024, 1, 32768, 1, 1024);
        tdm_load_2d(&sV[nb][0], Vbh + (kbase + KTILE), 16, 64, 512, 64, 512);
        __builtin_amdgcn_s_wait_tensorcnt((short)2);  // tile kt's pair done
      } else {
        __builtin_amdgcn_s_wait_tensorcnt((short)0);
      }
    }
    __syncthreads();                   // publish buf[cur] to all waves

    const _Float16* sKc = &sK[cur][0];
    const _Float16* sVc = &sV[cur][0];

    // ---- S = Q K^T : four 16x16 n-subtiles, d split in two 32-chunks ----
    v8f sc[4];
    #pragma unroll
    for (int ns = 0; ns < 4; ++ns) {
      v8f acc;
      #pragma unroll
      for (int i = 0; i < 8; ++i) acc[i] = 0.0f;
      #pragma unroll
      for (int g = 0; g < 2; ++g) {
        const v16h kb =
            *(const v16h*)&sKc[(16 * ns + laneLo) * DHEAD + 32 * g + 16 * laneHi];
        acc = __builtin_amdgcn_wmma_f32_16x16x32_f16(false, Qa[g], false, kb,
                                                     (short)0, acc, false, false);
      }
      sc[ns] = acc;
    }

    // ---- online softmax (mask only the diagonal tile; scalar branch) ----
    if (kt == nTiles - 1) {
      SOFTMAX_ROWS(true)
    } else {
      SOFTMAX_ROWS(false)
    }

    // ---- P as two 16x32 A-fragments (contiguous thanks to pi layout) ----
    v16h Pa[2];
    #pragma unroll
    for (int f = 0; f < 2; ++f)
      Pa[f] = *(const v16h*)&sPw[laneLo * KTILE + 32 * f + 16 * laneHi];

    // ---- O += P V : 4 d-subtiles x 2 key-fragments ----
    #pragma unroll
    for (int j = 0; j < 4; ++j) {
      #pragma unroll
      for (int f = 0; f < 2; ++f) {
        const _Float16* vs = &sVc[(16 * j + laneLo) * KTILE + 32 * f + 16 * laneHi];
        v8h lo = *(const v8h*)vs;
        v8h hi = *(const v8h*)(vs + 8);
        v16h vb;
        #pragma unroll
        for (int i = 0; i < 8; ++i) { vb[i] = lo[i]; vb[8 + i] = hi[i]; }
        Oacc[j] = __builtin_amdgcn_wmma_f32_16x16x32_f16(false, Pa[f], false, vb,
                                                         (short)0, Oacc[j],
                                                         false, false);
      }
    }

    __syncthreads();   // all reads of buf[cur] done before TDM overwrites it
  }

  #pragma unroll
  for (int r = 0; r < 8; ++r) {
    const float inv = 1.0f / lrow[r];
    float* orow = Ob + (size_t)(qrow0 + r + 8 * laneHi) * DHEAD;
    #pragma unroll
    for (int j = 0; j < 4; ++j)
      orow[16 * j + laneLo] = Oacc[j][r] * inv;
  }
}

// ============  fallback: fused kernel (no workspace), round-2 style  ========
__global__ __launch_bounds__(128)
void fa_fwd_fused(const float* __restrict__ Qg, const float* __restrict__ Kg,
                  const float* __restrict__ Vg, float* __restrict__ Og) {
  __shared__ __align__(32) _Float16 sK1[KTILE * DHEAD];
  __shared__ __align__(32) _Float16 sVt1[DHEAD * KTILE];
  __shared__ __align__(32) _Float16 sP[NWAVES][16 * KTILE];

  const int tid    = threadIdx.x;
  const int wave   = tid >> 5;
  const int lane   = tid & 31;
  const int laneLo = lane & 15;
  const int laneHi = lane >> 4;

  const int    q0     = blockIdx.x * QTILE;
  const size_t bh_off = (size_t)blockIdx.y * SLEN * DHEAD;
  const float* Qb = Qg + bh_off;
  const float* Kb = Kg + bh_off;
  const float* Vb = Vg + bh_off;
  float*       Ob = Og + bh_off;

  const int qrow0 = q0 + wave * 16;
  _Float16* sPw = &sP[wave][0];

  v16h Qa[2];
  {
    const float* qrow = Qb + (size_t)(qrow0 + laneLo) * DHEAD;
    #pragma unroll
    for (int f = 0; f < 2; ++f) {
      const int dbase = 32 * f + laneHi * 16;
      v4f a0 = *(const v4f*)(qrow + dbase);
      v4f a1 = *(const v4f*)(qrow + dbase + 4);
      v4f b0 = *(const v4f*)(qrow + dbase + 8);
      v4f b1 = *(const v4f*)(qrow + dbase + 12);
      #pragma unroll
      for (int i = 0; i < 4; ++i) {
        Qa[f][i]      = (_Float16)(a0[i] * SCALE_LOG2E);
        Qa[f][4 + i]  = (_Float16)(a1[i] * SCALE_LOG2E);
        Qa[f][8 + i]  = (_Float16)(b0[i] * SCALE_LOG2E);
        Qa[f][12 + i] = (_Float16)(b1[i] * SCALE_LOG2E);
      }
    }
  }

  v8f Oacc[4];
  #pragma unroll
  for (int j = 0; j < 4; ++j)
    #pragma unroll
    for (int i = 0; i < 8; ++i) Oacc[j][i] = 0.0f;

  float mrow[8], lrow[8];
  #pragma unroll
  for (int r = 0; r < 8; ++r) { mrow[r] = NEGBIG; lrow[r] = 0.0f; }

  const int nTiles = blockIdx.x + 1;
  const int srow = tid >> 1;
  const int scol = (tid & 1) * 32;
  const int vcol = 4 * (srow & 15) + (srow >> 4);

  for (int kt = 0; kt < nTiles; ++kt) {
    const int kbase = kt * KTILE;
    __syncthreads();
    {
      const float* kp = Kb + (size_t)(kbase + srow) * DHEAD + scol;
      const float* vp = Vb + (size_t)(kbase + srow) * DHEAD + scol;
      #pragma unroll
      for (int c = 0; c < 4; ++c) {
        v4f x0 = *(const v4f*)(kp + 8 * c);
        v4f x1 = *(const v4f*)(kp + 8 * c + 4);
        v8h h;
        #pragma unroll
        for (int i = 0; i < 4; ++i) { h[i] = (_Float16)x0[i]; h[4 + i] = (_Float16)x1[i]; }
        *(v8h*)&sK1[srow * DHEAD + scol + 8 * c] = h;
      }
      #pragma unroll
      for (int c = 0; c < 8; ++c) {
        v4f x = *(const v4f*)(vp + 4 * c);
        #pragma unroll
        for (int i = 0; i < 4; ++i)
          sVt1[(scol + 4 * c + i) * KTILE + vcol] = (_Float16)x[i];
      }
    }
    __syncthreads();

    v8f sc[4];
    #pragma unroll
    for (int ns = 0; ns < 4; ++ns) {
      v8f acc;
      #pragma unroll
      for (int i = 0; i < 8; ++i) acc[i] = 0.0f;
      #pragma unroll
      for (int g = 0; g < 2; ++g) {
        const v16h kb =
            *(const v16h*)&sK1[(16 * ns + laneLo) * DHEAD + 32 * g + 16 * laneHi];
        acc = __builtin_amdgcn_wmma_f32_16x16x32_f16(false, Qa[g], false, kb,
                                                     (short)0, acc, false, false);
      }
      sc[ns] = acc;
    }

    if (kt == nTiles - 1) {
      SOFTMAX_ROWS(true)
    } else {
      SOFTMAX_ROWS(false)
    }

    v16h Pa[2];
    #pragma unroll
    for (int f = 0; f < 2; ++f)
      Pa[f] = *(const v16h*)&sPw[laneLo * KTILE + 32 * f + 16 * laneHi];

    #pragma unroll
    for (int j = 0; j < 4; ++j) {
      #pragma unroll
      for (int f = 0; f < 2; ++f) {
        const _Float16* vs = &sVt1[(16 * j + laneLo) * KTILE + 32 * f + 16 * laneHi];
        v8h lo = *(const v8h*)vs;
        v8h hi = *(const v8h*)(vs + 8);
        v16h vb;
        #pragma unroll
        for (int i = 0; i < 8; ++i) { vb[i] = lo[i]; vb[8 + i] = hi[i]; }
        Oacc[j] = __builtin_amdgcn_wmma_f32_16x16x32_f16(false, Pa[f], false, vb,
                                                         (short)0, Oacc[j],
                                                         false, false);
      }
    }
  }

  #pragma unroll
  for (int r = 0; r < 8; ++r) {
    const float inv = 1.0f / lrow[r];
    float* orow = Ob + (size_t)(qrow0 + r + 8 * laneHi) * DHEAD;
    #pragma unroll
    for (int j = 0; j < 4; ++j)
      orow[16 * j + laneLo] = Oacc[j][r] * inv;
  }
}

extern "C" void kernel_launch(void* const* d_in, const int* in_sizes, int n_in,
                              void* d_out, int out_size, void* d_ws, size_t ws_size,
                              hipStream_t stream) {
  (void)in_sizes; (void)n_in; (void)out_size;
  const float* Q = (const float*)d_in[0];
  const float* K = (const float*)d_in[1];
  const float* V = (const float*)d_in[2];
  float*       O = (float*)d_out;

  dim3 grid(SLEN / QTILE, NBH);
  dim3 block(128);

  if (ws_size >= WS_NEEDED) {
    _Float16* Kh = (_Float16*)d_ws;
    _Float16* Vt = (_Float16*)((char*)d_ws + KH_BYTES);
    prep_kv<<<grid, block, 0, stream>>>(V, K, Kh, Vt);
    fa_fwd_tdm<<<grid, block, 0, stream>>>(Q, Kh, Vt, O);
  } else {
    fa_fwd_fused<<<grid, block, 0, stream>>>(Q, K, V, O);
  }
}
